// Relative_Temporal_SelfAttention__24979529793893
// MI455X (gfx1250) — matrix operand order repaired and checked by reading, not compiled, not run on hardware
//
#include <hip/hip_runtime.h>
#include <stdint.h>

// Relative Temporal Self-Attention, fused flash-attention style for gfx1250.
//   B=8, L=1024, D=512, H=8, DH=64
// Identity used: s_rel[b,h,i,j] = q[b,i,h,:] . We[L-1-|i-j|, :]
// Pipeline: f32->bf16 convert -> WMMA QKV GEMMs -> fused attention (QK^T +
// on-the-fly rel-term + online softmax + PV, all v_wmma_f32_16x16x32_bf16,
// K tiles staged with global_load_async_to_lds_b128 double-buffered)
// -> WMMA output GEMM with bias (f32 out).

#define B_  8
#define L_  1024
#define D_  512
#define H_  8
#define DH_ 64

typedef unsigned short bf16raw;
typedef __bf16   bf16x16 __attribute__((ext_vector_type(16)));
typedef float    f32x8   __attribute__((ext_vector_type(8)));
typedef uint32_t u32x8   __attribute__((ext_vector_type(8)));

union Frag {
  u32x8   u;
  bf16x16 h;
  f32x8   f;
};

static __device__ __forceinline__ bf16raw f2bf(float x) {
  union { float f; uint32_t u; } c; c.f = x;
  uint32_t u = c.u;
  u += 0x7fffu + ((u >> 16) & 1u);     // round-to-nearest-even
  return (bf16raw)(u >> 16);
}

static __device__ __forceinline__ f32x8 wmma_bf16(const Frag& a, const Frag& b, f32x8 c) {
  // D = A(16x32 bf16) * B(32x16 bf16) + C(16x16 f32)
  return __builtin_amdgcn_wmma_f32_16x16x32_bf16(false, a.h, false, b.h,
                                                 (short)0, c, false, false);
}

// Element offset (bf16 units, always even -> dword aligned) of the K-pair held
// in VGPR v of a 16x32 bf16 A fragment. Per ISA table: lanes 0-15 hold
// K={0..7,16..23}, lanes 16-31 hold K={8..15,24..31}, two K per VGPR.
static __device__ __forceinline__ int a_kpair(int v, int hi) {
  return ((v < 4) ? (2 * v) : (16 + 2 * (v - 4))) + (hi ? 8 : 0);
}
// Same for a 32x16 bf16 B fragment: same K order as A (lanes 0-15 hold
// K={0..7,16..23}, lanes 16-31 hold K={8..15,24..31}).
static __device__ __forceinline__ int b_kpair(int v, int hi) {
  return ((v < 4) ? (2 * v) : (16 + 2 * (v - 4))) + (hi ? 8 : 0);
}

// LDS byte offset of a __shared__ object: per ISA 10.2, the LDS aperture maps
// flat addr[31:0] directly onto the wave's LDS allocation, so the low 32 bits
// of the generic pointer are the ds-space offset the async ops expect.
static __device__ __forceinline__ uint32_t lds_off(const void* p) {
  return (uint32_t)(uintptr_t)p;
}

// ---------------------------------------------------------------------------
__global__ void cvt_f32_to_bf16_kernel(const float* __restrict__ src,
                                       bf16raw* __restrict__ dst, int n) {
  int i = blockIdx.x * blockDim.x + threadIdx.x;
  int stride = gridDim.x * blockDim.x;
  for (; i < n; i += stride) dst[i] = f2bf(src[i]);
}

// ---------------------------------------------------------------------------
// C(nrows x ncols, bf16) = A(nrows x K, bf16) @ W(ncols x K, bf16)^T
// One wave computes a 16x64 tile; all 5 fragments of a k-step are loaded as
// one clause before the 4 WMMAs so loads of step k+1 overlap WMMAs of step k.
__global__ __launch_bounds__(32) void gemm_bf16_nt_kernel(
    const bf16raw* __restrict__ A, const bf16raw* __restrict__ W,
    bf16raw* __restrict__ C, int K, int ncols) {
  const int lane = threadIdx.x;
  const int m  = lane & 15;
  const int hi = lane >> 4;
  const int row0 = blockIdx.x * 16;
  const int col0 = blockIdx.y * 64;

  f32x8 acc[4];
  for (int nt = 0; nt < 4; ++nt)
    for (int r = 0; r < 8; ++r) acc[nt][r] = 0.f;

  const bf16raw* arow = A + (size_t)(row0 + m) * K;
  for (int k0 = 0; k0 < K; k0 += 32) {
    if (k0 + 32 < K) __builtin_prefetch(arow + k0 + 32, 0, 1);
    Frag af, bfr[4];
#pragma unroll
    for (int v = 0; v < 8; ++v)
      af.u[v] = *(const uint32_t*)(arow + k0 + a_kpair(v, hi));
#pragma unroll
    for (int nt = 0; nt < 4; ++nt) {
      const bf16raw* wrow = W + (size_t)(col0 + nt * 16 + m) * K + k0;
#pragma unroll
      for (int v = 0; v < 8; ++v)
        bfr[nt].u[v] = *(const uint32_t*)(wrow + b_kpair(v, hi));
    }
#pragma unroll
    for (int nt = 0; nt < 4; ++nt)
      acc[nt] = wmma_bf16(af, bfr[nt], acc[nt]);
  }
#pragma unroll
  for (int nt = 0; nt < 4; ++nt)
#pragma unroll
    for (int r = 0; r < 8; ++r)
      C[(size_t)(row0 + hi * 8 + r) * ncols + col0 + nt * 16 + m] =
          f2bf(acc[nt][r]);
}

// ---------------------------------------------------------------------------
// out(f32) = A(bf16) @ W(bf16)^T + bias
__global__ __launch_bounds__(32) void gemm_out_kernel(
    const bf16raw* __restrict__ A, const bf16raw* __restrict__ W,
    const float* __restrict__ bias, float* __restrict__ C, int K, int ncols) {
  const int lane = threadIdx.x;
  const int m  = lane & 15;
  const int hi = lane >> 4;
  const int row0 = blockIdx.x * 16;
  const int col0 = blockIdx.y * 64;

  f32x8 acc[4];
  for (int nt = 0; nt < 4; ++nt)
    for (int r = 0; r < 8; ++r) acc[nt][r] = 0.f;

  const bf16raw* arow = A + (size_t)(row0 + m) * K;
  for (int k0 = 0; k0 < K; k0 += 32) {
    if (k0 + 32 < K) __builtin_prefetch(arow + k0 + 32, 0, 1);
    Frag af, bfr[4];
#pragma unroll
    for (int v = 0; v < 8; ++v)
      af.u[v] = *(const uint32_t*)(arow + k0 + a_kpair(v, hi));
#pragma unroll
    for (int nt = 0; nt < 4; ++nt) {
      const bf16raw* wrow = W + (size_t)(col0 + nt * 16 + m) * K + k0;
#pragma unroll
      for (int v = 0; v < 8; ++v)
        bfr[nt].u[v] = *(const uint32_t*)(wrow + b_kpair(v, hi));
    }
#pragma unroll
    for (int nt = 0; nt < 4; ++nt)
      acc[nt] = wmma_bf16(af, bfr[nt], acc[nt]);
  }
#pragma unroll
  for (int nt = 0; nt < 4; ++nt)
#pragma unroll
    for (int r = 0; r < 8; ++r) {
      const int col = col0 + nt * 16 + m;
      C[(size_t)(row0 + hi * 8 + r) * ncols + col] = acc[nt][r] + bias[col];
    }
}

// ---------------------------------------------------------------------------
// Staging helpers for the fused attention kernel (256 threads cooperate).

// K tile: 32 keys x 64 bf16 = 4 KB = one async b128 per thread straight into
// LDS (no VGPR round-trip, tracked with ASYNCcnt).
static __device__ __forceinline__ void issue_async_k(
    const bf16raw* __restrict__ kb, int b, int h, int j0,
    bf16raw* kdst, int tid) {
  const int j   = tid >> 3;            // key row (0..31)
  const int seg = tid & 7;             // 16-byte segment within row
  const uint64_t src = (uint64_t)(uintptr_t)(
      kb + ((size_t)(b * L_ + j0 + j)) * D_ + h * DH_ + seg * 8);
  const uint32_t dst = lds_off(kdst) + (uint32_t)(tid * 16);
  asm volatile("global_load_async_to_lds_b128 %0, %1, off"
               :: "v"(dst), "v"(src) : "memory");
}

// V tile is transposed into LDS ([d][key]) so it must go through VGPRs:
// issue the global loads early, commit the ds stores later (pipelined).
static __device__ __forceinline__ void issue_v_loads(
    const bf16raw* __restrict__ vb, int b, int h, int j0, int tid,
    uint32_t vv[2]) {
#pragma unroll
  for (int rep = 0; rep < 2; ++rep) {
    const int idx = tid + rep * 256;   // 0..511 = 32 keys x 16 dwords
    const int j   = idx >> 4;
    const int dw  = idx & 15;
    vv[rep] = *(const uint32_t*)(
        vb + ((size_t)(b * L_ + j0 + j)) * D_ + h * DH_ + dw * 2);
  }
}

static __device__ __forceinline__ void store_v_transposed(
    bf16raw* vtdst, int tid, const uint32_t vv[2]) {
#pragma unroll
  for (int rep = 0; rep < 2; ++rep) {
    const int idx = tid + rep * 256;
    const int j   = idx >> 4;
    const int dw  = idx & 15;
    vtdst[(dw * 2 + 0) * 32 + j] = (bf16raw)(vv[rep] & 0xffffu);
    vtdst[(dw * 2 + 1) * 32 + j] = (bf16raw)(vv[rep] >> 16);
  }
}

// ---------------------------------------------------------------------------
// Fused attention: one block = 8 waves, each wave owns one 16-query block of
// one (b,h); waves share double-buffered K/V tiles in LDS. Online softmax in
// exp2 domain; relative-position logits computed per 32-key tile via an extra
// 16x48 WMMA against We rows [e_lo, e_lo+48).
__global__ __launch_bounds__(256) void attn_fused_kernel(
    const bf16raw* __restrict__ qb, const bf16raw* __restrict__ kb,
    const bf16raw* __restrict__ vb, const bf16raw* __restrict__ web,
    bf16raw* __restrict__ ob) {
  __shared__ bf16raw kbuf[2][32 * 64];     // [buf][key][d]   8 KB
  __shared__ bf16raw vtbuf[2][64 * 32];    // [buf][d][key]   8 KB
  __shared__ float   prelbuf[8][16 * 48];  // per-wave rel logits, 24 KB
  __shared__ bf16raw pbuf[8][16 * 32];     // per-wave P staging,   8 KB

  const int tid  = threadIdx.x;
  const int wave = tid >> 5;
  const int lane = tid & 31;
  const int m    = lane & 15;
  const int hi   = lane >> 4;

  const int bid = blockIdx.x;              // 512 blocks
  const int bh  = bid >> 3;                // b*H + h
  const int b   = bh >> 3;                 // H_ == 8
  const int h   = bh & 7;
  const int i0  = ((bid & 7) * 8 + wave) * 16;

  // Q fragments for this wave's 16 queries, kept in registers (K = DH = 64).
  Frag qf[2];
  {
    const bf16raw* qrow = qb + ((size_t)(b * L_ + i0 + m)) * D_ + h * DH_;
#pragma unroll
    for (int ks = 0; ks < 2; ++ks)
#pragma unroll
      for (int v = 0; v < 8; ++v)
        qf[ks].u[v] = *(const uint32_t*)(qrow + ks * 32 + a_kpair(v, hi));
  }

  f32x8 oacc[4];
  for (int nt = 0; nt < 4; ++nt)
    for (int r = 0; r < 8; ++r) oacc[nt][r] = 0.f;
  float m_run[8], l_run[8];
  for (int r = 0; r < 8; ++r) { m_run[r] = -1e30f; l_run[r] = 0.f; }

  const float kscale = 0.125f * 1.4426950408889634f;  // 1/sqrt(DH) * log2(e)
  const int NT = L_ / 32;

  // ---- prologue: stage tile 0 into buffer 0 ----
  {
    issue_async_k(kb, b, h, 0, kbuf[0], tid);
    uint32_t vv0[2];
    issue_v_loads(vb, b, h, 0, tid, vv0);
    store_v_transposed(vtbuf[0], tid, vv0);
    asm volatile("s_wait_asynccnt 0x0" ::: "memory");
    __syncthreads();
  }

  for (int jt = 0; jt < NT; ++jt) {
    const int j0  = jt * 32;
    const int cur = jt & 1;
    const bool have_next = (jt + 1) < NT;

    // ---- pipeline: kick off next tile's K (async->LDS) and V (to regs) ----
    uint32_t vv[2];
    if (have_next) {
      issue_async_k(kb, b, h, j0 + 32, kbuf[cur ^ 1], tid);
      issue_v_loads(vb, b, h, j0 + 32, tid, vv);
    }

    // ---- rel-position logits: e = L-1-|i-j| spans <=47 contiguous values
    int dmin;
    if (j0 > i0 + 15)      dmin = j0 - (i0 + 15);
    else if (i0 > j0 + 31) dmin = i0 - (j0 + 31);
    else                   dmin = 0;
    const int e_max = (L_ - 1) - dmin;
    const int e_lo  = (e_max - 47 > 0) ? (e_max - 47) : 0;

#pragma unroll
    for (int et = 0; et < 3; ++et) {
      f32x8 pr;
      for (int r = 0; r < 8; ++r) pr[r] = 0.f;
      const bf16raw* wrow = web + (size_t)(e_lo + et * 16 + m) * DH_;
#pragma unroll
      for (int ks = 0; ks < 2; ++ks) {
        Frag ef;
#pragma unroll
        for (int v = 0; v < 8; ++v)
          ef.u[v] = *(const uint32_t*)(wrow + ks * 32 + b_kpair(v, hi));
        pr = wmma_bf16(qf[ks], ef, pr);
      }
#pragma unroll
      for (int r = 0; r < 8; ++r)
        prelbuf[wave][(hi * 8 + r) * 48 + et * 16 + m] = pr[r];
    }

    // ---- S = Q K^T (two 16x16 tiles over 32 keys) ----
    f32x8 sc[2];
#pragma unroll
    for (int nt = 0; nt < 2; ++nt) {
      for (int r = 0; r < 8; ++r) sc[nt][r] = 0.f;
      const int jl = nt * 16 + m;
#pragma unroll
      for (int ks = 0; ks < 2; ++ks) {
        Frag kf;
#pragma unroll
        for (int v = 0; v < 8; ++v)
          kf.u[v] =
              *(const uint32_t*)&kbuf[cur][jl * 64 + ks * 32 + b_kpair(v, hi)];
        sc[nt] = wmma_bf16(qf[ks], kf, sc[nt]);
      }
    }

    // ---- add rel term, scale into log2 domain ----
#pragma unroll
    for (int nt = 0; nt < 2; ++nt)
#pragma unroll
      for (int r = 0; r < 8; ++r) {
        const int row = hi * 8 + r;
        const int i   = i0 + row;
        const int j   = j0 + nt * 16 + m;
        const int ad  = (i > j) ? (i - j) : (j - i);
        const float rel = prelbuf[wave][row * 48 + ((L_ - 1 - ad) - e_lo)];
        sc[nt][r] = (sc[nt][r] + rel) * kscale;
      }

    // ---- online softmax (rows live in one 16-lane half; butterfly xor) ----
    float alpha[8];
#pragma unroll
    for (int r = 0; r < 8; ++r) {
      float mt = fmaxf(sc[0][r], sc[1][r]);
#pragma unroll
      for (int off = 1; off < 16; off <<= 1)
        mt = fmaxf(mt, __shfl_xor(mt, off, 32));
      const float mnew = fmaxf(m_run[r], mt);
      alpha[r] = exp2f(m_run[r] - mnew);
      const float p0 = exp2f(sc[0][r] - mnew);
      const float p1 = exp2f(sc[1][r] - mnew);
      sc[0][r] = p0; sc[1][r] = p1;
      float ps = p0 + p1;
#pragma unroll
      for (int off = 1; off < 16; off <<= 1)
        ps += __shfl_xor(ps, off, 32);
      l_run[r] = l_run[r] * alpha[r] + ps;
      m_run[r] = mnew;
    }
#pragma unroll
    for (int nt = 0; nt < 4; ++nt)
#pragma unroll
      for (int r = 0; r < 8; ++r) oacc[nt][r] *= alpha[r];

    // ---- commit next tile's V transpose (global latency now hidden) ----
    if (have_next) store_v_transposed(vtbuf[cur ^ 1], tid, vv);

    // ---- C-layout P -> A-layout bf16 via per-wave LDS, then O += P @ V ----
#pragma unroll
    for (int nt = 0; nt < 2; ++nt)
#pragma unroll
      for (int r = 0; r < 8; ++r)
        pbuf[wave][(hi * 8 + r) * 32 + nt * 16 + m] = f2bf(sc[nt][r]);

    Frag pf;
#pragma unroll
    for (int v = 0; v < 8; ++v)
      pf.u[v] = *(const uint32_t*)&pbuf[wave][m * 32 + a_kpair(v, hi)];

#pragma unroll
    for (int nt = 0; nt < 4; ++nt) {
      Frag vf;
      const int d = nt * 16 + m;
#pragma unroll
      for (int v = 0; v < 8; ++v)
        vf.u[v] = *(const uint32_t*)&vtbuf[cur][d * 32 + b_kpair(v, hi)];
      oacc[nt] = wmma_bf16(pf, vf, oacc[nt]);
    }

    // next tile's async K writes + everyone done reading current buffers
    asm volatile("s_wait_asynccnt 0x0" ::: "memory");
    __syncthreads();
  }

  // ---- normalize and store bf16 O ----
  float inv[8];
  for (int r = 0; r < 8; ++r) inv[r] = 1.f / l_run[r];
#pragma unroll
  for (int nt = 0; nt < 4; ++nt)
#pragma unroll
    for (int r = 0; r < 8; ++r) {
      const int row = hi * 8 + r;
      ob[((size_t)(b * L_ + i0 + row)) * D_ + h * DH_ + nt * 16 + m] =
          f2bf(oacc[nt][r] * inv[r]);
    }
}

// ---------------------------------------------------------------------------
extern "C" void kernel_launch(void* const* d_in, const int* in_sizes, int n_in,
                              void* d_out, int out_size, void* d_ws, size_t ws_size,
                              hipStream_t stream) {
  const float* x  = (const float*)d_in[0];
  const float* Wq = (const float*)d_in[1];
  const float* Wk = (const float*)d_in[2];
  const float* Wv = (const float*)d_in[3];
  const float* We = (const float*)d_in[4];
  const float* Wo = (const float*)d_in[5];
  const float* bo = (const float*)d_in[6];
  float* out = (float*)d_out;

  const size_t NX = (size_t)B_ * L_ * D_;  // 4,194,304
  const size_t NW = (size_t)D_ * D_;       // 262,144
  const size_t NE = (size_t)L_ * DH_;      // 65,536

  bf16raw* p = (bf16raw*)d_ws;
  bf16raw* xb  = p; p += NX;
  bf16raw* wqb = p; p += NW;
  bf16raw* wkb = p; p += NW;
  bf16raw* wvb = p; p += NW;
  bf16raw* wob = p; p += NW;
  bf16raw* web = p; p += NE;
  bf16raw* qbf = p; p += NX;
  bf16raw* kbf = p; p += NX;
  bf16raw* vbf = p; p += NX;
  bf16raw* obf = p; p += NX;   // ~42 MB total bf16 workspace

  // f32 -> bf16 conversions
  cvt_f32_to_bf16_kernel<<<(int)(NX / 1024), 256, 0, stream>>>(x,  xb,  (int)NX);
  cvt_f32_to_bf16_kernel<<<(int)(NW / 256),  256, 0, stream>>>(Wq, wqb, (int)NW);
  cvt_f32_to_bf16_kernel<<<(int)(NW / 256),  256, 0, stream>>>(Wk, wkb, (int)NW);
  cvt_f32_to_bf16_kernel<<<(int)(NW / 256),  256, 0, stream>>>(Wv, wvb, (int)NW);
  cvt_f32_to_bf16_kernel<<<(int)(NW / 256),  256, 0, stream>>>(Wo, wob, (int)NW);
  cvt_f32_to_bf16_kernel<<<(int)(NE / 256),  256, 0, stream>>>(We, web, (int)NE);

  // QKV projections: (B*L x D) @ (D x D)^T
  dim3 g1((B_ * L_) / 16, D_ / 64);
  gemm_bf16_nt_kernel<<<g1, 32, 0, stream>>>(xb, wqb, qbf, D_, D_);
  gemm_bf16_nt_kernel<<<g1, 32, 0, stream>>>(xb, wkb, kbf, D_, D_);
  gemm_bf16_nt_kernel<<<g1, 32, 0, stream>>>(xb, wvb, vbf, D_, D_);

  // Fused attention: 512 blocks x 8 waves, each wave one 16-query block
  attn_fused_kernel<<<B_ * H_ * (L_ / 16) / 8, 256, 0, stream>>>(
      qbf, kbf, vbf, web, obf);

  // Output projection + bias (f32 out)
  gemm_out_kernel<<<g1, 32, 0, stream>>>(obf, wob, bo, out, D_, D_);

  (void)in_sizes; (void)n_in; (void)out_size; (void)ws_size;
}
